// MultiHeadAttentionLayer_1821066134324
// MI455X (gfx1250) — compile-verified, matrix operand order checked by reading, not run on hardware
//
#include <hip/hip_runtime.h>
#include <hip/hip_bf16.h>

#define DEV __device__ __forceinline__

typedef __attribute__((ext_vector_type(16))) _Float16 v16h;
typedef __attribute__((ext_vector_type(8)))  float    v8f;

union OpU { uint4 u[2]; v16h v; };
union PkU { uint4 u; _Float16 h[8]; };
union I4U { int4 v; int a[4]; };

// Sizes fixed by the reference: B=2, S=2048, D=1024, H=16, DK=64, M=B*S=4096.

// Load one 16x32 f16 WMMA operand (A-format; also serves as B-format when the
// matrix is supplied as its transpose). rowp = &mat[this_lane_row*ld + col0].
// Lane half hh picks cols {8h..8h+7} and {16+8h..16+8h+7}: two b128 loads.
DEV v16h load_op16(const _Float16* rowp, int hh) {
  OpU o;
  o.u[0] = *(const uint4*)(rowp + 8 * hh);
  o.u[1] = *(const uint4*)(rowp + 16 + 8 * hh);
  return o.v;
}

DEV v8f wmma_f16(v16h a, v16h b, v8f c) {
  return __builtin_amdgcn_wmma_f32_16x16x32_f16(false, a, false, b, (short)0, c,
                                                false, false);
}

// ---------------- conversion kernels ----------------

__global__ void cvt_f16_kernel(const float* __restrict__ x,
                               _Float16* __restrict__ y, int n) {
  int i = blockIdx.x * 256 + threadIdx.x;
  if (i < n) y[i] = (_Float16)x[i];
}

// W stored [in(1024), out(1024)]; produce wT[out][in] in f16.
__global__ void cvt_wT_kernel(const float* __restrict__ W,
                              _Float16* __restrict__ wT) {
  int idx = blockIdx.x * 256 + threadIdx.x;  // 1M elements
  int o = idx >> 10, i = idx & 1023;
  wT[idx] = (_Float16)W[i * 1024 + o];
}

// ---------------- shared GEMM core: 64(M) x 64(N) per wave ----------------

DEV void gemm_load(const _Float16* arow, const _Float16* brow, int k, int hh,
                   v16h a[4], v16h b[4]) {
#pragma unroll
  for (int t = 0; t < 4; ++t) {
    a[t] = load_op16(arow + (size_t)t * 16 * 1024 + k, hh);
    b[t] = load_op16(brow + (size_t)t * 16 * 1024 + k, hh);
  }
}

DEV void gemm_mma(v8f c[4][4], const v16h a[4], const v16h b[4]) {
#pragma unroll
  for (int mt = 0; mt < 4; ++mt)
#pragma unroll
    for (int nt = 0; nt < 4; ++nt) c[mt][nt] = wmma_f16(a[mt], b[nt], c[mt][nt]);
}

// Full 64x64 accumulation with double-buffered operands so loads for k+32
// overlap the 16 WMMAs at k.
DEV void gemm_64x64(const _Float16* arow, const _Float16* brow, int hh,
                    v8f c[4][4]) {
  v16h a0[4], b0[4], a1[4], b1[4];
  gemm_load(arow, brow, 0, hh, a0, b0);
#pragma unroll 2
  for (int k = 0; k < 1024; k += 64) {
    gemm_load(arow, brow, k + 32, hh, a1, b1);
    gemm_mma(c, a0, b0);
    if (k + 64 < 1024) gemm_load(arow, brow, k + 64, hh, a0, b0);
    gemm_mma(c, a1, b1);
  }
}

// ---------------- q/k/v projection: Y = X*W + b ----------------
// Grid: (64 M-blocks, 16 heads). vmode==0: [b,h,s,dk]; vmode==1: V^T [b,h,dk,s].
__global__ __launch_bounds__(32) void proj_qkv_kernel(
    const _Float16* __restrict__ x16, const _Float16* __restrict__ wT,
    const float* __restrict__ bias, _Float16* __restrict__ out, int vmode) {
  const int lane = threadIdx.x;
  const int r = lane & 15, hh = lane >> 4;
  const int m0 = blockIdx.x << 6;   // [0,4096), 64-row block
  const int head = blockIdx.y;      // [0,16)
  const int n0 = head << 6;

  const _Float16* arow = x16 + (size_t)(m0 + r) * 1024;
  const _Float16* brow = wT + (size_t)(n0 + r) * 1024;
  float bv[4];
#pragma unroll
  for (int nt = 0; nt < 4; ++nt) bv[nt] = bias[n0 + 16 * nt + r];

  v8f c[4][4] = {};
  gemm_64x64(arow, brow, hh, c);

  const int bb = m0 >> 11;       // batch
  const int s0 = m0 & 2047;      // seq offset inside batch (64-aligned)
  if (vmode == 0) {
    _Float16* ob = out + ((size_t)(bb * 16 + head) * 2048 + s0) * 64;
#pragma unroll
    for (int mt = 0; mt < 4; ++mt)
#pragma unroll
      for (int i = 0; i < 8; ++i) {
        size_t off = (size_t)(16 * mt + i + 8 * hh) * 64 + r;
#pragma unroll
        for (int nt = 0; nt < 4; ++nt)
          ob[off + 16 * nt] = (_Float16)(c[mt][nt][i] + bv[nt]);
      }
  } else {
    // vT[((b*16+head)*64 + dk)*2048 + s]; dk = 16*nt + r (fixed per lane),
    // s = s0 + 16*mt + i + 8*hh contiguous over i -> b128 stores.
    _Float16* ob = out + (size_t)(bb * 16 + head) * 64 * 2048;
#pragma unroll
    for (int mt = 0; mt < 4; ++mt)
#pragma unroll
      for (int nt = 0; nt < 4; ++nt) {
        PkU p;
#pragma unroll
        for (int i = 0; i < 8; ++i) p.h[i] = (_Float16)(c[mt][nt][i] + bv[nt]);
        *(uint4*)(ob + (size_t)(16 * nt + r) * 2048 + s0 + 16 * mt + 8 * hh) =
            p.u;
      }
  }
}

// ---------------- output projection: out = attn*Wo + bo (f32) ----------------
__global__ __launch_bounds__(32) void proj_o_kernel(
    const _Float16* __restrict__ a16, const _Float16* __restrict__ wT,
    const float* __restrict__ bias, float* __restrict__ out) {
  const int lane = threadIdx.x;
  const int r = lane & 15, hh = lane >> 4;
  const int m0 = blockIdx.x << 6;
  const int n0 = blockIdx.y << 6;

  const _Float16* arow = a16 + (size_t)(m0 + r) * 1024;
  const _Float16* brow = wT + (size_t)(n0 + r) * 1024;
  float bv[4];
#pragma unroll
  for (int nt = 0; nt < 4; ++nt) bv[nt] = bias[n0 + 16 * nt + r];

  v8f c[4][4] = {};
  gemm_64x64(arow, brow, hh, c);

#pragma unroll
  for (int mt = 0; mt < 4; ++mt)
#pragma unroll
    for (int i = 0; i < 8; ++i) {
      float* orow = out + (size_t)(m0 + 16 * mt + i + 8 * hh) * 1024 + n0 + r;
#pragma unroll
      for (int nt = 0; nt < 4; ++nt) orow[16 * nt] = c[mt][nt][i] + bv[nt];
    }
}

// ---------------- flash attention ----------------
// One wave owns 64 queries (4 tiles) of one (b,head); K/V tile loads are
// shared across the 4 query tiles. Scores computed transposed (S^T = K*Q^T)
// so lane index == query; probabilities feed O^T = V^T * P^T in-register.
__global__ __launch_bounds__(32) void flash_attn_kernel(
    const _Float16* __restrict__ q16, const _Float16* __restrict__ k16,
    const _Float16* __restrict__ vT16, const int* __restrict__ mask,
    _Float16* __restrict__ attn) {
  const int lane = threadIdx.x;
  const int r = lane & 15, hh = lane >> 4;
  const int qblk = blockIdx.x & 31;           // S/64 = 32
  const int head = (blockIdx.x >> 5) & 15;
  const int bb = blockIdx.x >> 9;
  const int q0 = qblk << 6;
  const size_t bh = (size_t)(bb * 16 + head);

  v16h qB[4][2];
  const int* mrow[4];
#pragma unroll
  for (int qt = 0; qt < 4; ++qt) {
    const _Float16* qp = q16 + (bh * 2048 + q0 + 16 * qt + r) * 64;
    qB[qt][0] = load_op16(qp, hh);
    qB[qt][1] = load_op16(qp + 32, hh);
    mrow[qt] = mask + ((size_t)bb * 2048 + (q0 + 16 * qt + r)) * 2048;
  }
  const _Float16* kb = k16 + bh * 2048 * 64;
  const _Float16* vb = vT16 + bh * 64 * 2048;

  float mrun[4] = {-1e30f, -1e30f, -1e30f, -1e30f};
  float lrun[4] = {0.0f, 0.0f, 0.0f, 0.0f};
  v8f o[4][4] = {};

  for (int kbase = 0; kbase < 2048; kbase += 32) {
    // K operands: two 16-key tiles x two dk-chunks (shared by all 4 q-tiles)
    const _Float16* k0p = kb + (size_t)(kbase + r) * 64;
    const _Float16* k1p = kb + (size_t)(kbase + 16 + r) * 64;
    const v16h kA00 = load_op16(k0p, hh);
    const v16h kA01 = load_op16(k0p + 32, hh);
    const v16h kA10 = load_op16(k1p, hh);
    const v16h kA11 = load_op16(k1p + 32, hh);
    // V^T operands: 4 dk-tiles x 32 keys (shared by all 4 q-tiles)
    const _Float16* v0p = vb + (size_t)r * 2048 + kbase;
    v16h vA[4];
#pragma unroll
    for (int t = 0; t < 4; ++t)
      vA[t] = load_op16(v0p + (size_t)(16 * t) * 2048, hh);

#pragma unroll
    for (int qt = 0; qt < 4; ++qt) {
      v8f s0 = {}, s1 = {};
      s0 = wmma_f16(kA00, qB[qt][0], s0);
      s0 = wmma_f16(kA01, qB[qt][1], s0);
      s1 = wmma_f16(kA10, qB[qt][0], s1);
      s1 = wmma_f16(kA11, qB[qt][1], s1);

      // scale + mask; element i holds key kbase + i + 8*hh (+16 for s1)
      I4U m00, m01, m10, m11;
      m00.v = *(const int4*)(mrow[qt] + kbase + 8 * hh);
      m01.v = *(const int4*)(mrow[qt] + kbase + 8 * hh + 4);
      m10.v = *(const int4*)(mrow[qt] + kbase + 16 + 8 * hh);
      m11.v = *(const int4*)(mrow[qt] + kbase + 16 + 8 * hh + 4);

      float sc0[8], sc1[8];
      float bm = -1e30f;
#pragma unroll
      for (int i = 0; i < 8; ++i) {
        float v0 = s0[i] * 0.125f;
        float v1 = s1[i] * 0.125f;
        const int mk0 = (i < 4) ? m00.a[i] : m01.a[i - 4];
        const int mk1 = (i < 4) ? m10.a[i] : m11.a[i - 4];
        v0 = (mk0 == 0) ? -1e9f : v0;
        v1 = (mk1 == 0) ? -1e9f : v1;
        sc0[i] = v0;
        sc1[i] = v1;
        bm = fmaxf(bm, fmaxf(v0, v1));
      }
      bm = fmaxf(bm, __shfl_xor(bm, 16, 32));  // 32-key row max per query

      const float mnew = fmaxf(mrun[qt], bm);
      const float corr = __expf(mrun[qt] - mnew);
      float bs = 0.0f;
      v16h a;  // P tile in A-operand layout (same-lane repack, no shuffles)
#pragma unroll
      for (int i = 0; i < 8; ++i) {
        const float p0 = __expf(sc0[i] - mnew);
        const float p1 = __expf(sc1[i] - mnew);
        bs += p0 + p1;
        a[i] = (_Float16)p0;
        a[8 + i] = (_Float16)p1;
      }
      bs += __shfl_xor(bs, 16, 32);
      lrun[qt] = lrun[qt] * corr + bs;
      mrun[qt] = mnew;

#pragma unroll
      for (int t = 0; t < 4; ++t)
        o[qt][t] = wmma_f16(vA[t], a, o[qt][t] * corr);
    }
  }

  // normalize and store attn[b][s][head*64 + d] (f16, heads re-merged)
#pragma unroll
  for (int qt = 0; qt < 4; ++qt) {
    const float inv = 1.0f / lrun[qt];
    _Float16* ob =
        attn + ((size_t)bb * 2048 + q0 + 16 * qt + r) * 1024 + head * 64;
#pragma unroll
    for (int t = 0; t < 4; ++t) {
      PkU p;
#pragma unroll
      for (int i = 0; i < 8; ++i) p.h[i] = (_Float16)(o[qt][t][i] * inv);
      *(uint4*)(ob + 16 * t + 8 * hh) = p.u;
    }
  }
}

// ---------------- host side ----------------

extern "C" void kernel_launch(void* const* d_in, const int* in_sizes, int n_in,
                              void* d_out, int out_size, void* d_ws,
                              size_t ws_size, hipStream_t stream) {
  const float* query = (const float*)d_in[0];
  const float* key_ = (const float*)d_in[1];
  const float* value = (const float*)d_in[2];
  const int* mask = (const int*)d_in[3];
  const float* Wq = (const float*)d_in[4];
  const float* bq = (const float*)d_in[5];
  const float* Wo = (const float*)d_in[6];
  const float* bo = (const float*)d_in[7];
  const float* Wk = (const float*)d_in[8];
  const float* bk = (const float*)d_in[9];
  const float* Wv = (const float*)d_in[10];
  const float* bv = (const float*)d_in[11];
  float* out = (float*)d_out;

  const size_t NX = 4096u * 1024u;   // B*S*D
  const size_t NW = 1024u * 1024u;   // D*D

  _Float16* w = (_Float16*)d_ws;
  _Float16* x16q = w;                 // 3 converted activations
  _Float16* x16k = x16q + NX;
  _Float16* x16v = x16k + NX;
  _Float16* wTq = x16v + NX;          // 4 transposed weights
  _Float16* wTk = wTq + NW;
  _Float16* wTv = wTk + NW;
  _Float16* wTo = wTv + NW;
  _Float16* q16 = wTo + NW;           // [b,h,s,dk]
  _Float16* k16 = q16 + NX;           // [b,h,s,dk]
  _Float16* vT16 = k16 + NX;          // [b,h,dk,s]
  _Float16* attn16 = vT16 + NX;       // [b,s,d]

  // f32 -> f16 conversions
  cvt_f16_kernel<<<(int)(NX / 256), 256, 0, stream>>>(query, x16q, (int)NX);
  cvt_f16_kernel<<<(int)(NX / 256), 256, 0, stream>>>(key_, x16k, (int)NX);
  cvt_f16_kernel<<<(int)(NX / 256), 256, 0, stream>>>(value, x16v, (int)NX);
  cvt_wT_kernel<<<(int)(NW / 256), 256, 0, stream>>>(Wq, wTq);
  cvt_wT_kernel<<<(int)(NW / 256), 256, 0, stream>>>(Wk, wTk);
  cvt_wT_kernel<<<(int)(NW / 256), 256, 0, stream>>>(Wv, wTv);
  cvt_wT_kernel<<<(int)(NW / 256), 256, 0, stream>>>(Wo, wTo);

  // projections: 64x64 tile per wave; grid (64 M-blocks, 16 heads)
  dim3 pg(64, 16);
  proj_qkv_kernel<<<pg, 32, 0, stream>>>(x16q, wTq, bq, q16, 0);
  proj_qkv_kernel<<<pg, 32, 0, stream>>>(x16k, wTk, bk, k16, 0);
  proj_qkv_kernel<<<pg, 32, 0, stream>>>(x16v, wTv, bv, vT16, 1);

  // flash attention: B*H*(S/64) = 1024 waves, 64 queries each
  flash_attn_kernel<<<1024, 32, 0, stream>>>(q16, k16, vT16, mask, attn16);

  // output projection
  proj_o_kernel<<<pg, 32, 0, stream>>>(attn16, wTo, bo, out);
}
